// SelfAttentionHead_1382979469694
// MI455X (gfx1250) — matrix operand order repaired and checked by reading, not compiled, not run on hardware
//
#include <hip/hip_runtime.h>
#include <hip/hip_bf16.h>

// MI455X / gfx1250 self-attention head, f16 WMMA (f32 accum) with
// CDNA5 async global->LDS staging (ASYNCcnt), double buffering, and
// software-pipelined LDS fragment loads.
//
// ws layout (bytes):
//   [0, 32MB)        x16 : x as f16, row-major [B*S][1024]
//   [32MB, +384KB)   wT  : Wq,Wk,Wv as f16, transposed [3][64][1024]
//   next 3 x 2MB     q16 (pre-scaled 1/8), k16 row-major [B*S][64],
//                    vT transposed [B][64][S]

typedef __attribute__((ext_vector_type(4)))  _Float16 h4;
typedef __attribute__((ext_vector_type(8)))  _Float16 h8;
typedef __attribute__((ext_vector_type(16))) _Float16 h16;
typedef __attribute__((ext_vector_type(8)))  float    f8;

static constexpr int BB  = 8;
static constexpr int SS  = 2048;
static constexpr int DIN = 1024;
static constexpr int DH  = 64;

#define DEVFN static __device__ __forceinline__

DEVFN h16 frag16(const _Float16* p0, const _Float16* p1) {
  h8 lo = *(const h8*)p0;
  h8 hi = *(const h8*)p1;
  return __builtin_shufflevector(lo, hi, 0,1,2,3,4,5,6,7,8,9,10,11,12,13,14,15);
}

DEVFN f8 wmma_f16(h16 a, h16 b, f8 c) {
  return __builtin_amdgcn_wmma_f32_16x16x32_f16(false, a, false, b, (short)0, c,
                                                false, false);
}

DEVFN float rmax16(float v) {
  v = fmaxf(v, __shfl_xor(v, 1, 32));
  v = fmaxf(v, __shfl_xor(v, 2, 32));
  v = fmaxf(v, __shfl_xor(v, 4, 32));
  v = fmaxf(v, __shfl_xor(v, 8, 32));
  return v;
}
DEVFN float rsum16(float v) {
  v += __shfl_xor(v, 1, 32);
  v += __shfl_xor(v, 2, 32);
  v += __shfl_xor(v, 4, 32);
  v += __shfl_xor(v, 8, 32);
  return v;
}

// CDNA5 async DMA: copy 16 contiguous global bytes into LDS (ASYNCcnt).
DEVFN void async_b128(unsigned lds_byte_off, const void* gptr) {
  asm volatile("global_load_async_to_lds_b128 %0, %1, off"
               :: "v"(lds_byte_off), "v"((unsigned long long)(uintptr_t)gptr)
               : "memory");
}
#define WAIT_ASYNC(n) asm volatile("s_wait_asynccnt " #n ::: "memory")
DEVFN void wait_ds0() { asm volatile("s_wait_dscnt 0x0" ::: "memory"); }

// ---------------- stage 0: precision / layout conversion ----------------

__global__ __launch_bounds__(256) void cvt_x_kernel(const float* __restrict__ x,
                                                    _Float16* __restrict__ x16) {
  size_t i = ((size_t)blockIdx.x * blockDim.x + threadIdx.x) * 4;
  float4 v = *(const float4*)(x + i);
  h4 o;
  o[0] = (_Float16)v.x; o[1] = (_Float16)v.y;
  o[2] = (_Float16)v.z; o[3] = (_Float16)v.w;
  *(h4*)(x16 + i) = o;
}

__global__ __launch_bounds__(256) void cvt_w_kernel(const float* __restrict__ w,
                                                    _Float16* __restrict__ wT) {
  int i = blockIdx.x * blockDim.x + threadIdx.x;   // i = k*DH + n
  int k = i >> 6;
  int n = i & (DH - 1);
  wT[(size_t)n * DIN + k] = (_Float16)w[i];
}

// ---------------- stage 1: fused QKV projection ----------------
// Block = 8 waves = 8 row-tiles. W k-slice (3*64 rows x 32 k, f16) shared by
// all waves -> async-staged in LDS, double buffered; B-fragment ds loads are
// depth-2 software pipelined against the WMMAs.
static constexpr int WROW   = 80;                // bytes per padded W row
static constexpr int WSLICE = 192 * WROW;        // 15360 B per buffer

DEVFN void issue_wslice(unsigned buf_off, const _Float16* __restrict__ wT,
                        int k0, int tid) {
#pragma unroll
  for (int j = 0; j < 3; ++j) {                  // 768 chunks / 256 threads
    int c = tid + j * 256;
    int row = c >> 2, part = c & 3;
    async_b128(buf_off + row * WROW + part * 16,
               wT + (size_t)row * DIN + k0 + part * 8);
  }
}

__global__ __launch_bounds__(256) void qkv_proj_kernel(
    const _Float16* __restrict__ x16, const _Float16* __restrict__ wT,
    const float* __restrict__ bq, const float* __restrict__ bk,
    const float* __restrict__ bv,
    _Float16* __restrict__ q16, _Float16* __restrict__ k16,
    _Float16* __restrict__ vT) {
  extern __shared__ char smem[];                 // 2 * WSLICE
  const int tid  = threadIdx.x;
  const int lane = tid & 31;
  const int wv   = tid >> 5;
  const int tile = blockIdx.x * 8 + wv;
  const int g    = lane >> 4;
  const int ln   = lane & 15;
  const int row0 = tile * 16;

  f8 acc[12] = {};
  const _Float16* xrow = x16 + (size_t)(row0 + ln) * DIN;

  issue_wslice(0, wT, 0, tid);                   // prefetch slice 0
  h16 a = frag16(xrow + g * 8, xrow + g * 8 + 16);

  for (int k0 = 0; k0 < DIN; k0 += 32) {
    const int buf = (k0 >> 5) & 1;
    // Issue next slice's DMA first, then wait only for the previous slice
    // (async loads complete in order; 3 loads just issued stay outstanding).
    if (k0 + 32 < DIN) {
      issue_wslice((buf ^ 1) * WSLICE, wT, k0 + 32, tid);
      WAIT_ASYNC(0x3);
    } else {
      WAIT_ASYNC(0x0);
    }
    __syncthreads();                             // slice(buf) visible to all

    const int kn = (k0 + 32) & (DIN - 1);        // A double-buffer (wraps)
    h16 a_next = frag16(xrow + kn + g * 8, xrow + kn + g * 8 + 16);

    const char* wb = smem + buf * WSLICE;
    auto ldsfrag = [&](int i) -> h16 {
      const _Float16* wp = (const _Float16*)(
          wb + ((i >> 2) * 64 + (i & 3) * 16 + ln) * WROW + g * 16);
      return frag16(wp, wp + 16);
    };
    // depth-2 pipelined: fragment i+2 loads while fragment i feeds the MMA
    h16 b0 = ldsfrag(0), b1 = ldsfrag(1);
#pragma unroll
    for (int i = 0; i < 12; ++i) {
      h16 bn = ldsfrag(i + 2 < 12 ? i + 2 : 11);
      acc[i] = wmma_f16(a, b0, acc[i]);
      b0 = b1;
      b1 = bn;
    }
    __syncthreads();                             // done reading slice(buf)
    a = a_next;
  }

  const int   b   = row0 / SS;
  const int   sr0 = row0 % SS;
  const float qscale = 0.125f;                   // 1/sqrt(DH) folded into q
#pragma unroll
  for (int t = 0; t < 4; ++t) {
    const int col = t * 16 + ln;
    const float bqv = bq[col], bkv = bk[col], bvv = bv[col];
#pragma unroll
    for (int r = 0; r < 8; ++r) {
      const int row = row0 + r + 8 * g;
      q16[(size_t)row * DH + col] = (_Float16)((acc[t][r]     + bqv) * qscale);
      k16[(size_t)row * DH + col] = (_Float16)( acc[4 + t][r] + bkv);
    }
    h8 vp;
#pragma unroll
    for (int r = 0; r < 8; ++r) vp[r] = (_Float16)(acc[8 + t][r] + bvv);
    *(h8*)(vT + ((size_t)b * DH + col) * SS + sr0 + 8 * g) = vp;
  }
}

// ---------------- stage 2: flash attention ----------------
static constexpr int KROW   = 160;               // padded K row (64 f16 + pad)
static constexpr int VROW   = 80;                // padded V row (32 f16 + pad)
static constexpr int KSLICE = 32 * KROW;         // 5120 B
static constexpr int VSLICE = 64 * VROW;         // 5120 B
static constexpr int VBASE  = 2 * KSLICE;
static constexpr int PBASE  = VBASE + 2 * VSLICE;
static constexpr int ATTN_LDS = PBASE + 8 * 16 * 40 * 2;  // 30720 B

DEVFN void issue_kv(const _Float16* __restrict__ k16,
                    const _Float16* __restrict__ vT,
                    int b, int kc, int buf, int tid) {
  {  // K: 32 rows x 128B payload = 256 x 16B chunks
    int row = tid >> 3, part = tid & 7;
    async_b128(buf * KSLICE + row * KROW + part * 16,
               k16 + ((size_t)b * SS + kc + row) * DH + part * 8);
  }
  {  // V: 64 rows x 64B payload = 256 x 16B chunks
    int row = tid >> 2, part = tid & 3;
    async_b128(VBASE + buf * VSLICE + row * VROW + part * 16,
               vT + ((size_t)b * DH + row) * SS + kc + part * 8);
  }
}

__global__ __launch_bounds__(256) void attn_kernel(
    const _Float16* __restrict__ q16, const _Float16* __restrict__ k16,
    const _Float16* __restrict__ vT, float* __restrict__ out) {
  extern __shared__ char smem[];
  const int tid  = threadIdx.x;
  const int lane = tid & 31;
  const int wv   = tid >> 5;
  const int tile = blockIdx.x * 8 + wv;
  const int b    = tile / (SS / 16);
  const int qt   = tile % (SS / 16);
  const int g    = lane >> 4;
  const int ln   = lane & 15;
  const int q0   = qt * 16;

  _Float16* plds = (_Float16*)(smem + PBASE) + wv * 16 * 40;

  issue_kv(k16, vT, b, 0, 0, tid);               // prefetch chunk 0

  const _Float16* qb = q16 + ((size_t)b * SS + q0 + ln) * DH;
  const h16 aq0 = frag16(qb +      g * 8, qb +      g * 8 + 16);
  const h16 aq1 = frag16(qb + 32 + g * 8, qb + 32 + g * 8 + 16);

  f8 o[4] = {};
  float m[8], l[8];
#pragma unroll
  for (int r = 0; r < 8; ++r) { m[r] = -3.0e38f; l[r] = 0.0f; }

  for (int kc = 0; kc < SS; kc += 32) {
    const int buf = (kc >> 5) & 1;
    if (kc + 32 < SS) {                          // issue next DMA, then wait
      issue_kv(k16, vT, b, kc + 32, buf ^ 1, tid);
      WAIT_ASYNC(0x2);                           // only previous chunk's 2
    } else {
      WAIT_ASYNC(0x0);
    }
    __syncthreads();                             // chunk(buf) visible

    const char* kbuf = smem + buf * KSLICE;
    const char* vbuf = smem + VBASE + buf * VSLICE;

    // ---- all 4 K fragments up front, then the score WMMAs ----
    h16 bk[2][2];
#pragma unroll
    for (int sub = 0; sub < 2; ++sub) {
      const _Float16* kb =
          (const _Float16*)(kbuf + (sub * 16 + ln) * KROW + g * 16);
      bk[sub][0] = frag16(kb,      kb + 16);     // d 0..31
      bk[sub][1] = frag16(kb + 32, kb + 48);     // d 32..63
    }
    f8 s[2];
#pragma unroll
    for (int sub = 0; sub < 2; ++sub) {
      f8 c = {};
      c = wmma_f16(aq0, bk[sub][0], c);
      c = wmma_f16(aq1, bk[sub][1], c);
      s[sub] = c;
    }

    // ---- V fragments issued early: ds loads overlap softmax VALU ----
    h16 bv[4];
#pragma unroll
    for (int t = 0; t < 4; ++t) {
      const _Float16* vp =
          (const _Float16*)(vbuf + (t * 16 + ln) * VROW + g * 16);
      bv[t] = frag16(vp, vp + 16);
    }

    // ---- online softmax ----
    float corr[8];
#pragma unroll
    for (int r = 0; r < 8; ++r) {
      float mx = rmax16(fmaxf(s[0][r], s[1][r]));
      float mn = fmaxf(m[r], mx);
      corr[r]  = __expf(m[r] - mn);
      m[r]     = mn;
    }
#pragma unroll
    for (int r = 0; r < 8; ++r) {
      float p0 = __expf(s[0][r] - m[r]);
      float p1 = __expf(s[1][r] - m[r]);
      l[r] = l[r] * corr[r] + rsum16(p0 + p1);
#pragma unroll
      for (int t = 0; t < 4; ++t) o[t][r] *= corr[r];
      plds[(r + 8 * g) * 40 +      ln] = (_Float16)p0;
      plds[(r + 8 * g) * 40 + 16 + ln] = (_Float16)p1;
    }
    wait_ds0();                                  // per-wave P roundtrip

    const _Float16* pr = plds + ln * 40 + g * 8;
    h16 ap = frag16(pr, pr + 16);

#pragma unroll
    for (int t = 0; t < 4; ++t) o[t] = wmma_f16(ap, bv[t], o[t]);

    __syncthreads();                             // done reading chunk(buf)
  }

  float* ob = out + ((size_t)b * SS + q0) * DH;
#pragma unroll
  for (int t = 0; t < 4; ++t)
#pragma unroll
    for (int r = 0; r < 8; ++r) {
      const int row = r + 8 * g;
      ob[(size_t)row * DH + t * 16 + ln] = o[t][r] / l[r];
    }
}

// ---------------- host launcher ----------------

extern "C" void kernel_launch(void* const* d_in, const int* in_sizes, int n_in,
                              void* d_out, int out_size, void* d_ws,
                              size_t ws_size, hipStream_t stream) {
  const float* x  = (const float*)d_in[0];
  const float* Wq = (const float*)d_in[1];
  const float* bq = (const float*)d_in[2];
  const float* Wk = (const float*)d_in[3];
  const float* bk = (const float*)d_in[4];
  const float* Wv = (const float*)d_in[5];
  const float* bv = (const float*)d_in[6];
  float* out = (float*)d_out;

  char* ws = (char*)d_ws;
  const size_t x16_bytes = (size_t)BB * SS * DIN * 2;
  const size_t wT_bytes  = (size_t)3 * DH * DIN * 2;
  const size_t qkv_elems = (size_t)BB * SS * DH;

  _Float16* x16 = (_Float16*)ws;
  _Float16* wT  = (_Float16*)(ws + x16_bytes);
  _Float16* q16 = (_Float16*)(ws + x16_bytes + wT_bytes);
  _Float16* k16 = q16 + qkv_elems;
  _Float16* vT  = k16 + qkv_elems;

  cvt_x_kernel<<<(BB * SS * DIN / 4) / 256, 256, 0, stream>>>(x, x16);
  cvt_w_kernel<<<(DIN * DH) / 256, 256, 0, stream>>>(Wq, wT + 0 * (size_t)DH * DIN);
  cvt_w_kernel<<<(DIN * DH) / 256, 256, 0, stream>>>(Wk, wT + 1 * (size_t)DH * DIN);
  cvt_w_kernel<<<(DIN * DH) / 256, 256, 0, stream>>>(Wv, wT + 2 * (size_t)DH * DIN);

  qkv_proj_kernel<<<(BB * SS / 16) / 8, 256, 2 * WSLICE, stream>>>(
      x16, wT, bq, bk, bv, q16, k16, vT);
  attn_kernel<<<(BB * SS / 16) / 8, 256, ATTN_LDS, stream>>>(q16, k16, vT, out);
}
